// Main_53807350284568
// MI455X (gfx1250) — matrix operand order repaired and checked by reading, not compiled, not run on hardware
//
#include <hip/hip_runtime.h>

#define V_   128
#define H_   1024
#define B_   64
#define T_   256
#define G4H  4096
#define NBLK 64
#define NTHR 256

#define LDSROW 72               // 64 cols + 8 pad (bank-conflict-free ds_load_b128)
#define LDSBUF (64 * LDSROW)    // one staging buffer (elems)

typedef __attribute__((ext_vector_type(16))) __bf16 v16bf;
typedef __attribute__((ext_vector_type(8)))  __bf16 v8bf;
typedef __attribute__((ext_vector_type(8)))  float  v8f;

// ---------- helpers ----------
static __device__ __forceinline__ __bf16 to_bf16(float f) {
  union { float f; unsigned u; } v; v.f = f;
  unsigned r = v.u + 0x7FFFu + ((v.u >> 16) & 1u);   // round-to-nearest-even
  unsigned short h = (unsigned short)(r >> 16);
  __bf16 o; __builtin_memcpy(&o, &h, 2); return o;
}

static __device__ __forceinline__ float sigm(float x) { return 1.0f / (1.0f + __expf(-x)); }

static __device__ __forceinline__ v16bf mk16(v8bf lo, v8bf hi) {
  v16bf r;
#pragma unroll
  for (int i = 0; i < 8; i++) { r[i] = lo[i]; r[i + 8] = hi[i]; }
  return r;
}

// ---------- CDNA5 async global->LDS copy (ASYNCcnt-tracked) ----------
static __device__ __forceinline__ void async_ld128(unsigned ldsOff, const void* gp) {
  asm volatile("global_load_async_to_lds_b128 %0, %1, off"
               :: "v"(ldsOff), "v"((unsigned long long)(size_t)gp) : "memory");
}
static __device__ __forceinline__ void wait_async0() {
  asm volatile("s_wait_asynccnt 0x0" ::: "memory");
}

// ---------- register-only tile GEMM (used by the one-shot FC) ----------
static __device__ __forceinline__ void gemm_acc(
    v8f& acc, const __bf16* __restrict__ A, int lda,
    const __bf16* __restrict__ W, int ldw, int K,
    int m0, int n0, int lane16, int laneHalf)
{
  const __bf16* ap = A + (size_t)(m0 + lane16) * lda + laneHalf * 8;
  const __bf16* wp = W + (size_t)(n0 + lane16) * ldw + laneHalf * 8;
#pragma unroll 4
  for (int k = 0; k < K; k += 32) {
    v8bf alo = *(const v8bf*)(ap + k);
    v8bf ahi = *(const v8bf*)(ap + k + 16);
    v8bf blo = *(const v8bf*)(wp + k);
    v8bf bhi = *(const v8bf*)(wp + k + 16);
    acc = __builtin_amdgcn_wmma_f32_16x16x32_bf16(
        false, mk16(alo, ahi), false, mk16(blo, bhi), (short)0, acc, false, false);
  }
}

// ---------- async-staged strip GEMM for the recurrent gate matmuls ----------
// Block covers M=64 (all of B) x 64 columns of the gate matrix. Activation
// chunks A[0:64][k0:k0+64] stream into double-buffered LDS via
// global_load_async_to_lds_b128, one chunk ahead of compute; each wave computes
// 2 M-tiles for one N-tile (weight fragment reused, 2 independent WMMA chains).
static __device__ __forceinline__ void gemm_lds(
    v8f& acc0, v8f& acc1,
    const __bf16* __restrict__ A, int lda, int K,   // A: [64][lda]
    const __bf16* __restrict__ W, int ldw,          // W: [4H][ldw]
    int n0, int mh, int lane16, int laneHalf, int tid, __bf16* lds)
{
  const int row = tid >> 2;             // 0..63 : staging row
  const int seg = (tid & 3) << 4;       // 0,16,32,48 : staging col segment
  const __bf16* arow = A + (size_t)row * lda + seg;
  const __bf16* wp   = W + (size_t)(n0 + lane16) * ldw + laneHalf * 8;
  const __bf16* afb  = lds + (mh * 32 + lane16) * LDSROW + laneHalf * 8;
  const unsigned stBase = (unsigned)(size_t)(lds + row * LDSROW + seg); // LDS byte addr
  const int nch = K >> 6;

  __syncthreads();                      // previous LDS users finished
  async_ld128(stBase,      arow);       // chunk 0 -> buf 0
  async_ld128(stBase + 16, arow + 8);
  int p = 0;
  for (int c = 0; c < nch; ++c) {
    const int k0 = c << 6;
    wait_async0();                      // my chunk-c LDS writes landed
    __syncthreads();                    // everyone's landed; other buffer is free
    if (c + 1 < nch) {                  // stream chunk c+1 into the other buffer
      unsigned st = stBase + (unsigned)((p ^ 1) * LDSBUF * 2);
      async_ld128(st,      arow + k0 + 64);
      async_ld128(st + 16, arow + k0 + 64 + 8);
    }
    const __bf16* ab = afb + p * LDSBUF;
#pragma unroll
    for (int kk = 0; kk < 64; kk += 32) {
      v8bf blo = *(const v8bf*)(wp + k0 + kk);
      v8bf bhi = *(const v8bf*)(wp + k0 + kk + 16);
      v16bf bfr = mk16(blo, bhi);
      const __bf16* a0 = ab + kk;
      v16bf af0 = mk16(*(const v8bf*)(a0),                *(const v8bf*)(a0 + 16));
      v16bf af1 = mk16(*(const v8bf*)(a0 + 16 * LDSROW),  *(const v8bf*)(a0 + 16 * LDSROW + 16));
      acc0 = __builtin_amdgcn_wmma_f32_16x16x32_bf16(false, af0, false, bfr, (short)0, acc0, false, false);
      acc1 = __builtin_amdgcn_wmma_f32_16x16x32_bf16(false, af1, false, bfr, (short)0, acc1, false, false);
    }
    p ^= 1;
  }
}

// ---------- small prep kernels ----------
__global__ void cvt_f32_bf16(const float* __restrict__ s, __bf16* __restrict__ d, int n) {
  for (int i = blockIdx.x * blockDim.x + threadIdx.x; i < n; i += gridDim.x * blockDim.x)
    d[i] = to_bf16(s[i]);
}

__global__ void vec_add_f32(const float* __restrict__ a, const float* __restrict__ b,
                            float* __restrict__ o, int n) {
  for (int i = blockIdx.x * blockDim.x + threadIdx.x; i < n; i += gridDim.x * blockDim.x)
    o[i] = a[i] + b[i];
}

__global__ void copy_f32(const float* __restrict__ s, float* __restrict__ d, int n) {
  for (int i = blockIdx.x * blockDim.x + threadIdx.x; i < n; i += gridDim.x * blockDim.x)
    d[i] = s[i];
}

__global__ void init_bar(unsigned* bar) { if (threadIdx.x < 2) bar[threadIdx.x] = 0u; }

// ---------- device-wide barrier (sense-reversing, agent scope) ----------
static __device__ __forceinline__ void grid_sync(unsigned* cnt, unsigned* gen) {
  __syncthreads();
  __threadfence();
  if (threadIdx.x == 0) {
    unsigned g = __hip_atomic_load(gen, __ATOMIC_ACQUIRE, __HIP_MEMORY_SCOPE_AGENT);
    unsigned t = __hip_atomic_fetch_add(cnt, 1u, __ATOMIC_ACQ_REL, __HIP_MEMORY_SCOPE_AGENT);
    if (t == (unsigned)(NBLK - 1)) {
      __hip_atomic_store(cnt, 0u, __ATOMIC_RELAXED, __HIP_MEMORY_SCOPE_AGENT);
      __hip_atomic_fetch_add(gen, 1u, __ATOMIC_RELEASE, __HIP_MEMORY_SCOPE_AGENT);
    } else {
      while (__hip_atomic_load(gen, __ATOMIC_ACQUIRE, __HIP_MEMORY_SCOPE_AGENT) == g)
        __builtin_amdgcn_s_sleep(8);
    }
  }
  __syncthreads();
}

// ---------- persistent 2-layer LSTM over T steps ----------
__global__ void __launch_bounds__(NTHR) lstm_persistent(
    const __bf16* __restrict__ xbf,   // [T][B][V]
    const __bf16* __restrict__ WA1,   // [4H][V]
    const __bf16* __restrict__ WA2,   // [4H][H]
    const __bf16* __restrict__ WB1,   // [4H][H]
    const __bf16* __restrict__ WB2,   // [4H][H]
    const float*  __restrict__ bsumA, // [4H]
    const float*  __restrict__ bsumB, // [4H]
    __bf16* hAbf, __bf16* hBbf,       // [B][H] bf16 states (GEMM operands)
    float* cA, float* cB,             // [B][H] f32 cell states (aliased into d_out)
    float* hAf, float* hBf,           // [B][H] f32 hidden (aliased into d_out)
    float* gates,                     // [B][4H] scratch
    __bf16* outs,                     // [T][B][H] bf16, feeds final FC
    unsigned* bar)
{
  __shared__ __bf16 lds[2 * LDSBUF];  // 18 KB double-buffered staging

  const int lane     = threadIdx.x & 31;
  const int lane16   = lane & 15;
  const int laneHalf = lane >> 4;
  const int wid      = threadIdx.x >> 5;    // 0..7
  const int ntl      = wid & 3;             // N-tile within block
  const int mh       = wid >> 2;            // 0..1 : M half (2 M-tiles each)
  const int n0       = blockIdx.x * 64 + ntl * 16;
  const int tid      = blockIdx.x * NTHR + threadIdx.x;
  const int ltid     = threadIdx.x;
  unsigned* cnt = bar;
  unsigned* gen = bar + 1;

  for (int t = 0; t < T_; t++) {
    // ---- layer A gates: x_t @ WA1^T + hA @ WA2^T ----
    {
      v8f acc0 = {}, acc1 = {};
      const __bf16* xt = xbf + (size_t)t * B_ * V_;
      gemm_lds(acc0, acc1, xt,   V_, V_, WA1, V_, n0, mh, lane16, laneHalf, ltid, lds);
      gemm_lds(acc0, acc1, hAbf, H_, H_, WA2, H_, n0, mh, lane16, laneHalf, ltid, lds);
#pragma unroll
      for (int r = 0; r < 8; r++) {
        gates[(size_t)(mh * 32 +      laneHalf * 8 + r) * G4H + n0 + lane16] = acc0[r];
        gates[(size_t)(mh * 32 + 16 + laneHalf * 8 + r) * G4H + n0 + lane16] = acc1[r];
      }
    }
    grid_sync(cnt, gen);

    // ---- layer A cell update ----
    for (int i = tid; i < B_ * H_; i += NBLK * NTHR) {
      int b = i >> 10, h = i & (H_ - 1);
      const float* gr = gates + (size_t)b * G4H;
      float gi = gr[h]          + bsumA[h];
      float gf = gr[H_ + h]     + bsumA[H_ + h];
      float gc = gr[2 * H_ + h] + bsumA[2 * H_ + h];
      float go = gr[3 * H_ + h] + bsumA[3 * H_ + h];
      float c  = cA[i];
      float cn = sigm(gf) * c + sigm(gi) * tanhf(gc);
      float hn = sigm(go) * tanhf(cn);
      cA[i] = cn; hAf[i] = hn; hAbf[i] = to_bf16(hn);
    }
    grid_sync(cnt, gen);

    // ---- layer B gates: hA @ WB1^T + hB @ WB2^T ----
    {
      v8f acc0 = {}, acc1 = {};
      gemm_lds(acc0, acc1, hAbf, H_, H_, WB1, H_, n0, mh, lane16, laneHalf, ltid, lds);
      gemm_lds(acc0, acc1, hBbf, H_, H_, WB2, H_, n0, mh, lane16, laneHalf, ltid, lds);
#pragma unroll
      for (int r = 0; r < 8; r++) {
        gates[(size_t)(mh * 32 +      laneHalf * 8 + r) * G4H + n0 + lane16] = acc0[r];
        gates[(size_t)(mh * 32 + 16 + laneHalf * 8 + r) * G4H + n0 + lane16] = acc1[r];
      }
    }
    grid_sync(cnt, gen);

    // ---- layer B cell update + emit out_t ----
    for (int i = tid; i < B_ * H_; i += NBLK * NTHR) {
      int b = i >> 10, h = i & (H_ - 1);
      const float* gr = gates + (size_t)b * G4H;
      float gi = gr[h]          + bsumB[h];
      float gf = gr[H_ + h]     + bsumB[H_ + h];
      float gc = gr[2 * H_ + h] + bsumB[2 * H_ + h];
      float go = gr[3 * H_ + h] + bsumB[3 * H_ + h];
      float c  = cB[i];
      float cn = sigm(gf) * c + sigm(gi) * tanhf(gc);
      float hn = sigm(go) * tanhf(cn);
      cB[i] = cn; hBf[i] = hn;
      __bf16 hb = to_bf16(hn);
      hBbf[i] = hb;
      outs[(size_t)t * B_ * H_ + i] = hb;
    }
    grid_sync(cnt, gen);
  }
}

// ---------- final FC: [T*B, H] @ Wfc^T + bfc -> [T*B, V] ----------
__global__ void __launch_bounds__(NTHR) fc_kernel(
    const __bf16* __restrict__ outs,  // [T*B][H]
    const __bf16* __restrict__ Wfc,   // [V][H]
    const float*  __restrict__ bfc,   // [V]
    float* __restrict__ out)          // [T*B][V]
{
  const int lane     = threadIdx.x & 31;
  const int lane16   = lane & 15;
  const int laneHalf = lane >> 4;
  const int wave     = (blockIdx.x * NTHR + threadIdx.x) >> 5;  // 0..8191
  const int m0 = (wave >> 3) * 16;   // 1024 M-tiles
  const int n0 = (wave & 7) * 16;    // 8 N-tiles
  v8f acc = {};
  gemm_acc(acc, outs, H_, Wfc, H_, H_, m0, n0, lane16, laneHalf);
  float bias = bfc[n0 + lane16];
#pragma unroll
  for (int r = 0; r < 8; r++)
    out[(size_t)(m0 + laneHalf * 8 + r) * V_ + n0 + lane16] = acc[r] + bias;
}

// ---------- host ----------
extern "C" void kernel_launch(void* const* d_in, const int* in_sizes, int n_in,
                              void* d_out, int out_size, void* d_ws, size_t ws_size,
                              hipStream_t stream)
{
  (void)in_sizes; (void)n_in; (void)out_size; (void)ws_size;
  const float* x   = (const float*)d_in[0];
  const float* hA0 = (const float*)d_in[1];
  const float* cA0 = (const float*)d_in[2];
  const float* hB0 = (const float*)d_in[3];
  const float* cB0 = (const float*)d_in[4];
  const float* WA1 = (const float*)d_in[5];
  const float* bA1 = (const float*)d_in[6];
  const float* WA2 = (const float*)d_in[7];
  const float* bA2 = (const float*)d_in[8];
  const float* WB1 = (const float*)d_in[9];
  const float* bB1 = (const float*)d_in[10];
  const float* WB2 = (const float*)d_in[11];
  const float* bB2 = (const float*)d_in[12];
  const float* Wfc = (const float*)d_in[13];
  const float* bfc = (const float*)d_in[14];

  char* ws = (char*)d_ws;
  auto aln = [](size_t v) { return (v + 255) & ~(size_t)255; };
  size_t o = 0;
  __bf16* xbf   = (__bf16*)(ws + o); o = aln(o + (size_t)T_ * B_ * V_ * 2);
  __bf16* WA1b  = (__bf16*)(ws + o); o = aln(o + (size_t)G4H * V_ * 2);
  __bf16* WA2b  = (__bf16*)(ws + o); o = aln(o + (size_t)G4H * H_ * 2);
  __bf16* WB1b  = (__bf16*)(ws + o); o = aln(o + (size_t)G4H * H_ * 2);
  __bf16* WB2b  = (__bf16*)(ws + o); o = aln(o + (size_t)G4H * H_ * 2);
  __bf16* Wfcb  = (__bf16*)(ws + o); o = aln(o + (size_t)V_ * H_ * 2);
  __bf16* hAbf  = (__bf16*)(ws + o); o = aln(o + (size_t)B_ * H_ * 2);
  __bf16* hBbf  = (__bf16*)(ws + o); o = aln(o + (size_t)B_ * H_ * 2);
  float*  bsA   = (float*)(ws + o);  o = aln(o + (size_t)G4H * 4);
  float*  bsB   = (float*)(ws + o);  o = aln(o + (size_t)G4H * 4);
  float*  gates = (float*)(ws + o);  o = aln(o + (size_t)B_ * G4H * 4);
  __bf16* outs  = (__bf16*)(ws + o); o = aln(o + (size_t)T_ * B_ * H_ * 2);
  unsigned* bar = (unsigned*)(ws + o);

  // d_out layout: [fc out | hA | cA | hB | cB]; write states in place.
  float* out_fc = (float*)d_out;
  float* out_hA = out_fc + (size_t)T_ * B_ * V_;
  float* out_cA = out_hA + (size_t)B_ * H_;
  float* out_hB = out_cA + (size_t)B_ * H_;
  float* out_cB = out_hB + (size_t)B_ * H_;

  // convert everything to bf16 (row-major, K contiguous)
  cvt_f32_bf16<<<2048, 256, 0, stream>>>(x,   xbf,  T_ * B_ * V_);
  cvt_f32_bf16<<<1024, 256, 0, stream>>>(WA1, WA1b, G4H * V_);
  cvt_f32_bf16<<<2048, 256, 0, stream>>>(WA2, WA2b, G4H * H_);
  cvt_f32_bf16<<<2048, 256, 0, stream>>>(WB1, WB1b, G4H * H_);
  cvt_f32_bf16<<<2048, 256, 0, stream>>>(WB2, WB2b, G4H * H_);
  cvt_f32_bf16<<<256,  256, 0, stream>>>(Wfc, Wfcb, V_ * H_);
  cvt_f32_bf16<<<128,  256, 0, stream>>>(hA0, hAbf, B_ * H_);
  cvt_f32_bf16<<<128,  256, 0, stream>>>(hB0, hBbf, B_ * H_);
  vec_add_f32<<<16, 256, 0, stream>>>(bA1, bA2, bsA, G4H);
  vec_add_f32<<<16, 256, 0, stream>>>(bB1, bB2, bsB, G4H);
  copy_f32<<<64, 256, 0, stream>>>(cA0, out_cA, B_ * H_);
  copy_f32<<<64, 256, 0, stream>>>(cB0, out_cB, B_ * H_);
  init_bar<<<1, 32, 0, stream>>>(bar);

  lstm_persistent<<<NBLK, NTHR, 0, stream>>>(
      xbf, WA1b, WA2b, WB1b, WB2b, bsA, bsB,
      hAbf, hBbf, out_cA, out_cB, out_hA, out_hB,
      gates, outs, bar);

  fc_kernel<<<1024, NTHR, 0, stream>>>(outs, Wfcb, bfc, out_fc);
}
